// MegablocksMoEMLP_80333068304449
// MI455X (gfx1250) — matrix operand order repaired and checked by reading, not compiled, not run on hardware
//
#include <hip/hip_runtime.h>
#include <hip/hip_bf16.h>

typedef __attribute__((ext_vector_type(16))) __bf16 v16bf;
typedef __attribute__((ext_vector_type(8)))  float  v8f;

#define H_DIM 1024
#define F_DIM 4096
#define E_NUM 8
#define T_NUM 4096
#define LDK   40   // padded LDS stride in halves (80B rows, 16B-aligned)

__device__ __forceinline__ unsigned short f2bf(float f) {
    unsigned int u = __float_as_uint(f);
    unsigned int r = u + 0x7FFFu + ((u >> 16) & 1u);
    return (unsigned short)(r >> 16);
}
__device__ __forceinline__ unsigned int pk2bf(float a, float b) {
    return (unsigned int)f2bf(a) | ((unsigned int)f2bf(b) << 16);
}

union FragU { uint4 u[2]; v16bf v; };

// ---------------------------------------------------------------------------
// Kernel 1: router — scores = sigmoid(x @ rw), top-2, build per-expert lists
// ---------------------------------------------------------------------------
__global__ __launch_bounds__(256)
void router_kernel(const float* __restrict__ x, const float* __restrict__ rw,
                   int* __restrict__ counts, int* __restrict__ tlist,
                   float* __restrict__ gates)
{
    const int t = blockIdx.x;
    const int tid = threadIdx.x;
    __shared__ float part[256][8];

    float acc[8];
#pragma unroll
    for (int e = 0; e < 8; ++e) acc[e] = 0.f;

    const float4 xv = ((const float4*)(x + (size_t)t * H_DIM))[tid];
    const float* r0 = rw + (size_t)(4 * tid) * E_NUM;
#pragma unroll
    for (int j = 0; j < 4; ++j) {
        float xs = (&xv.x)[j];
#pragma unroll
        for (int e = 0; e < 8; ++e) acc[e] += xs * r0[j * 8 + e];
    }
#pragma unroll
    for (int e = 0; e < 8; ++e) part[tid][e] = acc[e];
    __syncthreads();
    for (int s = 128; s > 0; s >>= 1) {
        if (tid < s) {
#pragma unroll
            for (int e = 0; e < 8; ++e) part[tid][e] += part[tid + s][e];
        }
        __syncthreads();
    }
    if (tid == 0) {
        float sc[8];
#pragma unroll
        for (int e = 0; e < 8; ++e) sc[e] = 1.f / (1.f + expf(-part[0][e]));
        int b0 = 0;
        for (int e = 1; e < 8; ++e) if (sc[e] > sc[b0]) b0 = e;   // ties -> lower idx (jax)
        int b1 = -1;
        for (int e = 0; e < 8; ++e) {
            if (e == b0) continue;
            if (b1 < 0 || sc[e] > sc[b1]) b1 = e;
        }
        int p0 = atomicAdd(&counts[b0], 1);
        tlist[b0 * T_NUM + p0] = 2 * t;
        gates[2 * t] = sc[b0];
        int p1 = atomicAdd(&counts[b1], 1);
        tlist[b1 * T_NUM + p1] = 2 * t + 1;
        gates[2 * t + 1] = sc[b1];
    }
}

// ---------------------------------------------------------------------------
// Kernel 2: h = gelu(gather(x) @ w1[e] + b1[e])  -> bf16 rows per slot
// block = 128 threads (4 waves), tile 128 tokens x 64 F, K-step 32
// each wave: 2 A-frags x 4 B-frags -> 8 wmma per K step (B reuse x2)
// ---------------------------------------------------------------------------
__global__ __launch_bounds__(128)
void gemm1_kernel(const float* __restrict__ x, const float* __restrict__ w1,
                  const float* __restrict__ b1, const int* __restrict__ counts,
                  const int* __restrict__ tlist, unsigned short* __restrict__ hws)
{
    const int e  = blockIdx.z;
    const int rt = blockIdx.y;
    const int ft = blockIdx.x;
    const int cnt = counts[e];
    if (rt * 128 >= cnt) return;

    __shared__ __align__(16) unsigned short aT[128 * LDK]; // [m][k]
    __shared__ __align__(16) unsigned short bT[64 * LDK];  // [n][k] (B transposed)
    __shared__ int slots[128];

    const int tid = threadIdx.x;
    {
        int pos = rt * 128 + tid;
        slots[tid] = (pos < cnt) ? tlist[e * T_NUM + pos] : -1;
    }
    __syncthreads();

    const int lane = tid & 31;
    const int wv   = tid >> 5;
    const int f0   = ft * 64;

    v8f c[2][4];
#pragma unroll
    for (int mb = 0; mb < 2; ++mb)
#pragma unroll
        for (int n = 0; n < 4; ++n)
#pragma unroll
            for (int r = 0; r < 8; ++r) c[mb][n][r] = 0.f;

    for (int k0 = 0; k0 < H_DIM; k0 += 32) {
        // stage A: gathered x rows, fp32 -> packed bf16, uint2 LDS stores
#pragma unroll
        for (int cc = 0; cc < 8; ++cc) {
            int idx = tid + cc * 128;            // 1024 float4 slots
            int m = idx >> 3;
            int k = (idx & 7) * 4;
            int s = slots[m];
            int tok = (s >= 0) ? (s >> 1) : 0;
            float4 v = *(const float4*)(x + (size_t)tok * H_DIM + k0 + k);
            uint2 pk;
            pk.x = pk2bf(v.x, v.y);
            pk.y = pk2bf(v.z, v.w);
            *(uint2*)&aT[m * LDK + k] = pk;
        }
        // stage B^T: w1[e][k0+k..k+3][f0+n] -> bT[n][k..k+3] (packed)
#pragma unroll
        for (int cc = 0; cc < 4; ++cc) {
            int idx = tid + cc * 128;            // 512 slots: 64 n x 8 k-quads
            int n = idx & 63;
            int k = (idx >> 6) * 4;
            const float* gp = w1 + ((size_t)e * H_DIM + k0 + k) * F_DIM + f0 + n;
            float a0 = gp[0];
            float a1 = gp[F_DIM];
            float a2 = gp[2 * F_DIM];
            float a3 = gp[3 * F_DIM];
            uint2 pk;
            pk.x = pk2bf(a0, a1);
            pk.y = pk2bf(a2, a3);
            *(uint2*)&bT[n * LDK + k] = pk;
        }
        __syncthreads();

        FragU af[2];
#pragma unroll
        for (int mb = 0; mb < 2; ++mb) {
            int m = wv * 32 + mb * 16 + (lane & 15);
            const unsigned short* p = &aT[m * LDK + ((lane < 16) ? 0 : 8)];
            af[mb].u[0] = *(const uint4*)(p);
            af[mb].u[1] = *(const uint4*)(p + 16);
        }
#pragma unroll
        for (int ns = 0; ns < 4; ++ns) {
            FragU bfr;
            int n = ns * 16 + (lane & 15);
            const unsigned short* p = &bT[n * LDK + ((lane < 16) ? 0 : 8)];
            bfr.u[0] = *(const uint4*)(p);
            bfr.u[1] = *(const uint4*)(p + 16);
#pragma unroll
            for (int mb = 0; mb < 2; ++mb) {
                c[mb][ns] = __builtin_amdgcn_wmma_f32_16x16x32_bf16(
                    false, af[mb].v, false, bfr.v, (short)0, c[mb][ns], false, false);
            }
        }
        __syncthreads();
    }

    // epilogue: bias + exact-erf gelu, scatter bf16 rows by slot
#pragma unroll
    for (int ns = 0; ns < 4; ++ns) {
        int n = f0 + ns * 16 + (lane & 15);
        float bias = b1[e * F_DIM + n];
#pragma unroll
        for (int mb = 0; mb < 2; ++mb) {
#pragma unroll
            for (int r = 0; r < 8; ++r) {
                int mrow = wv * 32 + mb * 16 + r + ((lane < 16) ? 0 : 8);
                int s = slots[mrow];
                if (s >= 0) {
                    float v = c[mb][ns][r] + bias;
                    float g = 0.5f * v * (1.0f + erff(v * 0.70710678118654752f));
                    hws[(size_t)s * F_DIM + n] = f2bf(g);
                }
            }
        }
    }
}

// ---------------------------------------------------------------------------
// Kernel 3: y = gather(h) @ w2[e] + b2[e]  -> fp32 rows per slot
// A tile (already bf16) staged with global_load_async_to_lds_b64 (ASYNCcnt)
// ---------------------------------------------------------------------------
__global__ __launch_bounds__(128)
void gemm2_kernel(const unsigned short* __restrict__ hws, const float* __restrict__ w2,
                  const float* __restrict__ b2, const int* __restrict__ counts,
                  const int* __restrict__ tlist, float* __restrict__ yws,
                  const float* __restrict__ zpad)
{
    const int e  = blockIdx.z;
    const int rt = blockIdx.y;
    const int ht = blockIdx.x;
    const int cnt = counts[e];
    if (rt * 128 >= cnt) return;

    __shared__ __align__(16) unsigned short aT[128 * LDK];
    __shared__ __align__(16) unsigned short bT[64 * LDK];
    __shared__ int slots[128];

    const int tid = threadIdx.x;
    {
        int pos = rt * 128 + tid;
        slots[tid] = (pos < cnt) ? tlist[e * T_NUM + pos] : -1;
    }
    __syncthreads();

    const int lane = tid & 31;
    const int wv   = tid >> 5;
    const int h0   = ht * 64;

    v8f c[2][4];
#pragma unroll
    for (int mb = 0; mb < 2; ++mb)
#pragma unroll
        for (int n = 0; n < 4; ++n)
#pragma unroll
            for (int r = 0; r < 8; ++r) c[mb][n][r] = 0.f;

    for (int k0 = 0; k0 < F_DIM; k0 += 32) {
        // stage A: gathered bf16 h rows, async DMA directly into LDS
#pragma unroll
        for (int cc = 0; cc < 8; ++cc) {
            int idx = tid + cc * 128;            // 1024 8-byte slots
            int m = idx >> 3;
            int k = (idx & 7) * 4;
            int s = slots[m];
            const void* gp = (s >= 0)
                ? (const void*)(hws + (size_t)s * F_DIM + k0 + k)
                : (const void*)zpad;             // zeroed region for padded rows
            unsigned ldsoff = (unsigned)(size_t)&aT[m * LDK + k];
            asm volatile("global_load_async_to_lds_b64 %0, %1, off"
                         :: "v"(ldsoff), "v"(gp)
                         : "memory");
        }
        // stage B^T: w2[e][k0+k..k+3][h0+n] -> bT[n][k..k+3] (packed)
#pragma unroll
        for (int cc = 0; cc < 4; ++cc) {
            int idx = tid + cc * 128;
            int n = idx & 63;
            int k = (idx >> 6) * 4;
            const float* gp = w2 + ((size_t)e * F_DIM + k0 + k) * H_DIM + h0 + n;
            float a0 = gp[0];
            float a1 = gp[H_DIM];
            float a2 = gp[2 * H_DIM];
            float a3 = gp[3 * H_DIM];
            uint2 pk;
            pk.x = pk2bf(a0, a1);
            pk.y = pk2bf(a2, a3);
            *(uint2*)&bT[n * LDK + k] = pk;
        }
        asm volatile("s_wait_asynccnt 0" ::: "memory");
        __syncthreads();

        FragU af[2];
#pragma unroll
        for (int mb = 0; mb < 2; ++mb) {
            int m = wv * 32 + mb * 16 + (lane & 15);
            const unsigned short* p = &aT[m * LDK + ((lane < 16) ? 0 : 8)];
            af[mb].u[0] = *(const uint4*)(p);
            af[mb].u[1] = *(const uint4*)(p + 16);
        }
#pragma unroll
        for (int ns = 0; ns < 4; ++ns) {
            FragU bfr;
            int n = ns * 16 + (lane & 15);
            const unsigned short* p = &bT[n * LDK + ((lane < 16) ? 0 : 8)];
            bfr.u[0] = *(const uint4*)(p);
            bfr.u[1] = *(const uint4*)(p + 16);
#pragma unroll
            for (int mb = 0; mb < 2; ++mb) {
                c[mb][ns] = __builtin_amdgcn_wmma_f32_16x16x32_bf16(
                    false, af[mb].v, false, bfr.v, (short)0, c[mb][ns], false, false);
            }
        }
        __syncthreads();
    }

#pragma unroll
    for (int ns = 0; ns < 4; ++ns) {
        int n = h0 + ns * 16 + (lane & 15);
        float bias = b2[e * H_DIM + n];
#pragma unroll
        for (int mb = 0; mb < 2; ++mb) {
#pragma unroll
            for (int r = 0; r < 8; ++r) {
                int mrow = wv * 32 + mb * 16 + r + ((lane < 16) ? 0 : 8);
                int s = slots[mrow];
                if (s >= 0) yws[(size_t)s * H_DIM + n] = c[mb][ns][r] + bias;
            }
        }
    }
}

// ---------------------------------------------------------------------------
// Kernel 4: out[t] = g0 * y[2t] + g1 * y[2t+1]
// ---------------------------------------------------------------------------
__global__ __launch_bounds__(256)
void combine_kernel(const float* __restrict__ yws, const float* __restrict__ gates,
                    float* __restrict__ out)
{
    const int t = blockIdx.x;
    const int i = threadIdx.x;
    const float g0 = gates[2 * t];
    const float g1 = gates[2 * t + 1];
    const float4 a = ((const float4*)(yws + (size_t)(2 * t) * H_DIM))[i];
    const float4 b = ((const float4*)(yws + (size_t)(2 * t + 1) * H_DIM))[i];
    float4 o;
    o.x = g0 * a.x + g1 * b.x;
    o.y = g0 * a.y + g1 * b.y;
    o.z = g0 * a.z + g1 * b.z;
    o.w = g0 * a.w + g1 * b.w;
    ((float4*)(out + (size_t)t * H_DIM))[i] = o;
}

// ---------------------------------------------------------------------------
extern "C" void kernel_launch(void* const* d_in, const int* in_sizes, int n_in,
                              void* d_out, int out_size, void* d_ws, size_t ws_size,
                              hipStream_t stream)
{
    const float* x   = (const float*)d_in[0];   // [2,2048,1024]
    const float* rw  = (const float*)d_in[1];   // [1024,8]
    const float* w1  = (const float*)d_in[2];   // [8,1024,4096]
    const float* b1  = (const float*)d_in[3];   // [8,4096]
    const float* w2  = (const float*)d_in[4];   // [8,4096,1024]
    const float* b2  = (const float*)d_in[5];   // [8,1024]
    float* out = (float*)d_out;

    char* ws = (char*)d_ws;
    int*   counts = (int*)(ws + 0);                               // 32 B
    const float* zpad = (const float*)(ws + 64);                  // 192 B zeroed pad
    int*   tlist  = (int*)(ws + 256);                             // 8*4096*4 = 128 KB
    float* gates  = (float*)(ws + 256 + 131072);                  // 8192*4 = 32 KB
    unsigned short* hws = (unsigned short*)(ws + 164096);         // 8192*4096*2 = 64 MB
    float* yws    = (float*)(ws + 67272960);                      // 8192*1024*4 = 32 MB

    hipMemsetAsync(ws, 0, 256, stream);                           // counts + zpad

    router_kernel<<<T_NUM, 256, 0, stream>>>(x, rw, counts, tlist, gates);
    gemm1_kernel<<<dim3(F_DIM / 64, T_NUM / 128, E_NUM), 128, 0, stream>>>(
        x, w1, b1, counts, tlist, hws);
    gemm2_kernel<<<dim3(H_DIM / 64, T_NUM / 128, E_NUM), 128, 0, stream>>>(
        hws, w2, b2, counts, tlist, yws, zpad);
    combine_kernel<<<T_NUM, 256, 0, stream>>>(yws, gates, out);
}
